// RNNlayer_9448928051455
// MI455X (gfx1250) — matrix operand order repaired and checked by reading, not compiled, not run on hardware
//
#include <hip/hip_runtime.h>

#define BATCH 128
#define SEQT  1024
#define HDIM  512
#define IDIM  64

#define MS 4      // H slices (workgroups per batch-tile group)
#define MH 128    // H rows per slice
#define BT 8      // batch tiles
#define BN 16     // batch cols per tile (WMMA N)

#define WH_STRIDE  520   // padded LDS row stride (f16 elems) for W_hidden slice
#define WIN_STRIDE 72    // padded LDS row stride for W_in slice

#define C1 0.9f   // 1 - DT/TAU
#define C2 0.1f   // DT/TAU

typedef __attribute__((ext_vector_type(16))) _Float16 v16h;
typedef __attribute__((ext_vector_type(8)))  float    v8f;
typedef unsigned long long u64;

union Frag  { uint4 q[2]; v16h v; };
union Pack8 { _Float16 h[8]; uint4 q; };

__global__ __launch_bounds__(256) void cvt_f32_f16(const float* __restrict__ src,
                                                   _Float16* __restrict__ dst, int n) {
  int i = blockIdx.x * 256 + threadIdx.x;
  if (i < n) dst[i] = (_Float16)src[i];
}

__global__ void zero_i32(int* __restrict__ p, int n) {
  int i = threadIdx.x;
  if (i < n) p[i] = 0;
}

// Persistent-state RNN scan.
// grid = 32 blocks: blockIdx = ms*BT + bt.  block = 256 threads = 8 wave32.
// Wave w owns the 16x16 state tile: H rows [ms*128 + w*16, +16), B cols [bt*16, +16).
__global__ __launch_bounds__(256) void rnn_scan(
    const _Float16* __restrict__ whf,   // [H][H]   f16
    const _Float16* __restrict__ winf,  // [H][I]   f16
    const _Float16* __restrict__ xf,    // [B][T][I] f16
    const float*    __restrict__ sigma, // [H][T][B] f32
    const float*    __restrict__ h0,    // [H][B]   f32
    float*          __restrict__ out,   // [B][T][H] f32
    float*          __restrict__ hlast, // [H][B]   f32
    _Float16*       __restrict__ sbuf,  // [2][BT][BN][H] f16 (sigmoid exchange, dbl-buffered)
    int*            __restrict__ counters) // BT groups * 16 ints (cacheline spread)
{
  extern __shared__ _Float16 lds[];
  _Float16* WhL  = lds;                  // [MH][WH_STRIDE]
  _Float16* WinL = lds + MH * WH_STRIDE; // [MH][WIN_STRIDE]

  const int bt   = blockIdx.x & (BT - 1);
  const int ms   = blockIdx.x / BT;
  const int tid  = threadIdx.x;
  const int w    = tid >> 5;       // wave id 0..7
  const int lane = tid & 31;
  const int cn   = lane & 15;      // column / row-within-tile selector
  const int hi   = lane >> 4;      // lane half

  // ---- one-time LDS preload of this slice's weights (8B vector copies) ----
  {
    const u64* src = (const u64*)(whf + (size_t)ms * MH * HDIM);
    for (int idx = tid; idx < MH * (HDIM / 4); idx += 256) {
      int r = idx >> 7, c = idx & 127;
      *(u64*)(WhL + r * WH_STRIDE + c * 4) = src[r * (HDIM / 4) + c];
    }
    const u64* s2 = (const u64*)(winf + (size_t)ms * MH * IDIM);
    for (int idx = tid; idx < MH * (IDIM / 4); idx += 256) {
      int r = idx >> 4, c = idx & 15;
      *(u64*)(WinL + r * WIN_STRIDE + c * 4) = s2[r * (IDIM / 4) + c];
    }
  }
  __syncthreads();

  const int b0    = bt * BN;
  const int rowg  = ms * MH + w * 16 + hi * 8; // global H row base of this lane's C elems
  const int col   = b0 + cn;                   // global batch column of this lane
  const int aRowL = w * 16 + cn;               // LDS row for A-fragment reads
  const int akoff = hi * 8;                    // A-fragment K offset per lane half
  const int bkoff = hi * 8;                    // B-fragment K offset per lane half
  int* cnt = counters + bt * 16;

  // persistent hidden state in registers (WMMA C/D layout)
  float h[8];
#pragma unroll
  for (int r = 0; r < 8; ++r) h[r] = h0[(size_t)(rowg + r) * BATCH + col];

  for (int t = 0; t < SEQT; ++t) {
    _Float16* sb = sbuf + ((size_t)(t & 1) * BT + bt) * (BN * HDIM); // [16][512]

    // ---- sigmoid(h) -> packed f16 fragment -> staging buffer (B-fragment layout) ----
    Pack8 pk;
#pragma unroll
    for (int r = 0; r < 8; ++r) {
      float sv = 1.0f / (1.0f + __expf(-h[r]));
      pk.h[r] = (_Float16)sv;
    }
    *(uint4*)(sb + cn * HDIM + rowg) = pk.q;

    // ---- 4-way barrier among WGs sharing this batch tile ----
    __threadfence();
    __syncthreads();
    if (tid == 0) {
      __hip_atomic_fetch_add(cnt, 1, __ATOMIC_RELEASE, __HIP_MEMORY_SCOPE_AGENT);
      const int target = MS * (t + 1);
      while (__hip_atomic_load(cnt, __ATOMIC_ACQUIRE, __HIP_MEMORY_SCOPE_AGENT) < target)
        __builtin_amdgcn_s_sleep(1);
    }
    __syncthreads();

    v8f acc = {};

    // ---- input projection: W_in_slice[16x64] @ x_t^T[64x16] (A from LDS, B from global f16 x) ----
    {
      const _Float16* xb = xf + ((size_t)col * SEQT + t) * IDIM + bkoff;
      const _Float16* ap = WinL + aRowL * WIN_STRIDE + akoff;
#pragma unroll
      for (int kb = 0; kb < IDIM; kb += 32) {
        Frag a, b;
        a.q[0] = *(const uint4*)(ap + kb);
        a.q[1] = *(const uint4*)(ap + kb + 16);
        b.q[0] = *(const uint4*)(xb + kb);
        b.q[1] = *(const uint4*)(xb + kb + 16);
        acc = __builtin_amdgcn_wmma_f32_16x16x32_f16(false, a.v, false, b.v,
                                                     (short)0, acc, false, false);
      }
    }

    // ---- recurrent GEMM: W_hidden_slice[16x512] @ sigmoid(h)[512x16], K-loop of 16 WMMAs ----
    {
      const _Float16* bb = sb + cn * HDIM + bkoff;
      const _Float16* ap = WhL + aRowL * WH_STRIDE + akoff;
#pragma unroll 4
      for (int kb = 0; kb < HDIM; kb += 32) {
        Frag a, b;
        a.q[0] = *(const uint4*)(ap + kb);
        a.q[1] = *(const uint4*)(ap + kb + 16);
        b.q[0] = *(const uint4*)(bb + kb);
        b.q[1] = *(const uint4*)(bb + kb + 16);
        acc = __builtin_amdgcn_wmma_f32_16x16x32_f16(false, a.v, false, b.v,
                                                     (short)0, acc, false, false);
      }
    }

    // ---- leaky integration + emit out[b, t, :] ----
    const float* sgp = sigma + ((size_t)rowg * SEQT + t) * BATCH + col;
#pragma unroll
    for (int r = 0; r < 8; ++r) {
      float sg = sgp[(size_t)r * SEQT * BATCH];
      h[r] = C1 * h[r] + C2 * (acc[r] + sg);
    }
    float* op = out + ((size_t)col * SEQT + t) * HDIM + rowg;
    *(float4*)op       = make_float4(h[0], h[1], h[2], h[3]);
    *(float4*)(op + 4) = make_float4(h[4], h[5], h[6], h[7]);
  }

  // ---- h_last [H][B] ----
#pragma unroll
  for (int r = 0; r < 8; ++r)
    hlast[(size_t)(rowg + r) * BATCH + col] = h[r];
}

extern "C" void kernel_launch(void* const* d_in, const int* in_sizes, int n_in,
                              void* d_out, int out_size, void* d_ws, size_t ws_size,
                              hipStream_t stream) {
  (void)in_sizes; (void)n_in; (void)out_size; (void)ws_size;
  const float* x     = (const float*)d_in[0]; // [B,T,I]
  const float* W_in  = (const float*)d_in[1]; // [H,I]
  const float* W_h   = (const float*)d_in[2]; // [H,H]
  const float* sigma = (const float*)d_in[3]; // [H,T,B]
  const float* h0    = (const float*)d_in[4]; // [H,B]
  float* out   = (float*)d_out;
  float* hlast = out + (size_t)BATCH * SEQT * HDIM;

  char* ws = (char*)d_ws;
  size_t off = 0;
  _Float16* whf  = (_Float16*)(ws + off); off += (size_t)HDIM * HDIM * 2;         // 512 KB
  _Float16* winf = (_Float16*)(ws + off); off += (size_t)HDIM * IDIM * 2;         //  64 KB
  _Float16* xf   = (_Float16*)(ws + off); off += (size_t)BATCH * SEQT * IDIM * 2; //  16 MB
  _Float16* sbuf = (_Float16*)(ws + off); off += (size_t)2 * BT * BN * HDIM * 2;  // 256 KB
  int* counters  = (int*)(ws + off);      off += (size_t)BT * 16 * sizeof(int);

  const int nWh  = HDIM * HDIM;
  const int nWin = HDIM * IDIM;
  const int nX   = BATCH * SEQT * IDIM;
  cvt_f32_f16<<<(nWh  + 255) / 256, 256, 0, stream>>>(W_h,  whf,  nWh);
  cvt_f32_f16<<<(nWin + 255) / 256, 256, 0, stream>>>(W_in, winf, nWin);
  cvt_f32_f16<<<(nX   + 255) / 256, 256, 0, stream>>>(x,    xf,   nX);
  zero_i32<<<1, 256, 0, stream>>>(counters, BT * 16);

  const size_t shmem = (size_t)(MH * WH_STRIDE + MH * WIN_STRIDE) * sizeof(_Float16); // ~148 KB
  hipFuncSetAttribute((const void*)rnn_scan,
                      hipFuncAttributeMaxDynamicSharedMemorySize, (int)shmem);
  rnn_scan<<<dim3(MS * BT), dim3(256), shmem, stream>>>(
      whf, winf, xf, sigma, h0, out, hlast, sbuf, counters);
}
